// SqueezeExcitationLayer_37125697306889
// MI455X (gfx1250) — matrix operand order repaired and checked by reading, not compiled, not run on hardware
//
#include <hip/hip_runtime.h>
#include <hip/hip_bf16.h>

typedef __attribute__((ext_vector_type(16))) _Float16 v16h;
typedef __attribute__((ext_vector_type(8)))  float    v8f;
typedef __attribute__((ext_vector_type(4)))  float    v4f;

#define B_   64
#define C_   256
#define CR_  16
#define HW_  3136   // 56*56
#define HW4_ 784    // HW_/4

// ---------------------------------------------------------------------------
// Kernel 1: global average pool. One 256-thread block per (b,c) plane.
// Streams the plane with 128-bit loads, LDS tree reduction.
// ---------------------------------------------------------------------------
__global__ __launch_bounds__(256) void se_pool_kernel(const float* __restrict__ in,
                                                      float* __restrict__ pooled) {
    const int bc  = blockIdx.x;           // 0 .. B_*C_-1
    const int tid = threadIdx.x;          // 0 .. 255
    const v4f* plane = (const v4f*)(in + (size_t)bc * HW_);
    float s = 0.f;
    for (int i = tid; i < HW4_; i += 256) {
        v4f v = plane[i];                 // global_load_b128
        s += v.x + v.y + v.z + v.w;
    }
    __shared__ float red[256];
    red[tid] = s;
    __syncthreads();
    for (int off = 128; off > 0; off >>= 1) {
        if (tid < off) red[tid] += red[tid + off];
        __syncthreads();
    }
    if (tid == 0) pooled[bc] = red[0] * (1.0f / (float)HW_);
}

// ---------------------------------------------------------------------------
// Kernel 2: SE gate MLP via WMMA. Single block, 4 waves (wave32).
// Wave w owns batch rows [16w, 16w+16).
//   GEMM1: H = relu(pooled[64,256] @ w1^T[256,16] + b1)   -> 8x wmma K-steps
//   GEMM2: G = sigmoid(H[64,16] @ w2^T[16,256] + b2)      -> 16 N-tiles, K padded
// VGPR layouts per CDNA5 ISA 7.12.2:
//   A (16x32 f16): lane row = lane%16; lanes 0-15: a[0..7]=K0..7, a[8..15]=K16..23
//                  lanes 16-31: a[0..7]=K8..15, a[8..15]=K24..31
//   B (32x16 f16): lane col = lane%16; b[i] = B[k][col], k as for a[i] above
//   C (16x16 f32): c[r] -> row r + 8*(lane/16), col lane%16
// ---------------------------------------------------------------------------
__global__ __launch_bounds__(128) void se_gate_kernel(const float* __restrict__ pooled,
                                                      const float* __restrict__ w1,
                                                      const float* __restrict__ b1,
                                                      const float* __restrict__ w2,
                                                      const float* __restrict__ b2,
                                                      float* __restrict__ gate) {
    const int tid  = threadIdx.x;         // 0..127
    const int wave = tid >> 5;            // M-tile id, 0..3
    const int lane = tid & 31;
    const int hlf  = lane >> 4;           // 0: lanes 0-15, 1: lanes 16-31
    const int ln16 = lane & 15;

    __shared__ _Float16 lds_h[4][16][16];

    // ---- GEMM1: rows 16*wave .. 16*wave+15, K = 256 in 8 steps of 32
    const int row = 16 * wave + ln16;
    const float* arow = pooled + row * C_;
    const float* bcol = w1 + ln16 * C_;   // B[k][n] = w1[n*C_ + k], n = ln16
    v8f acc = {};
    #pragma unroll
    for (int kb = 0; kb < C_; kb += 32) {
        v16h a, b;
        #pragma unroll
        for (int i = 0; i < 8; ++i) {
            a[i]     = (_Float16)arow[kb + 8 * hlf + i];
            a[i + 8] = (_Float16)arow[kb + 16 + 8 * hlf + i];
        }
        #pragma unroll
        for (int i = 0; i < 16; ++i) {
            b[i] = (_Float16)bcol[kb + 8 * hlf + i + (i >= 8 ? 8 : 0)];
        }
        acc = __builtin_amdgcn_wmma_f32_16x16x32_f16(false, a, false, b,
                                                     (short)0, acc, false, false);
    }
    // bias + ReLU, stash H tile in LDS as f16
    {
        const float bias1 = b1[ln16];
        #pragma unroll
        for (int r = 0; r < 8; ++r) {
            float h = acc[r] + bias1;
            h = h > 0.f ? h : 0.f;
            lds_h[wave][r + 8 * hlf][ln16] = (_Float16)h;
        }
    }
    __syncthreads();

    // ---- GEMM2: A = H tile (K=16, zero-padded to 32)
    v16h a2;
    #pragma unroll
    for (int i = 0; i < 8; ++i) {
        a2[i]     = lds_h[wave][ln16][8 * hlf + i];  // K = 8*hlf + i  (0..15)
        a2[i + 8] = (_Float16)0.f;                   // K = 16..31 padded
    }
    for (int nt = 0; nt < 16; ++nt) {
        v16h bv;
        #pragma unroll
        for (int i = 0; i < 16; ++i) bv[i] = (_Float16)0.f;
        {                                            // k = 8*hlf + i (i<8); k>=16 is pad
            const float* w2row = w2 + (nt * 16 + ln16) * CR_;  // w2[n][k]
            #pragma unroll
            for (int i = 0; i < 8; ++i) bv[i] = (_Float16)w2row[8 * hlf + i];
        }
        v8f c2 = {};
        c2 = __builtin_amdgcn_wmma_f32_16x16x32_f16(false, a2, false, bv,
                                                    (short)0, c2, false, false);
        const int ncol  = nt * 16 + ln16;
        const float bias2 = b2[ncol];
        #pragma unroll
        for (int r = 0; r < 8; ++r) {
            const int m = 16 * wave + r + 8 * hlf;
            float g = c2[r] + bias2;
            g = 1.0f / (1.0f + __expf(-g));
            gate[m * C_ + ncol] = g;
        }
    }
}

// ---------------------------------------------------------------------------
// Kernel 3: channel-wise rescale. One block per (b,c) plane.
// b128 loads, nontemporal b128 stores (output never re-read).
// ---------------------------------------------------------------------------
__global__ __launch_bounds__(256) void se_scale_kernel(const float* __restrict__ in,
                                                       const float* __restrict__ gate,
                                                       float* __restrict__ out) {
    const int bc  = blockIdx.x;
    const int tid = threadIdx.x;
    const float g = gate[bc];
    const v4f* ip = (const v4f*)(in + (size_t)bc * HW_);
    v4f*       op = (v4f*)(out + (size_t)bc * HW_);
    for (int i = tid; i < HW4_; i += 256) {
        v4f v = ip[i];
        v *= g;
        __builtin_nontemporal_store(v, op + i);
    }
}

extern "C" void kernel_launch(void* const* d_in, const int* in_sizes, int n_in,
                              void* d_out, int out_size, void* d_ws, size_t ws_size,
                              hipStream_t stream) {
    const float* in = (const float*)d_in[0];   // [64,256,56,56]
    const float* w1 = (const float*)d_in[1];   // [16,256]
    const float* b1 = (const float*)d_in[2];   // [16]
    const float* w2 = (const float*)d_in[3];   // [256,16]
    const float* b2 = (const float*)d_in[4];   // [256]
    float* out = (float*)d_out;

    float* pooled = (float*)d_ws;              // 64*256 floats
    float* gate   = pooled + B_ * C_;          // 64*256 floats

    se_pool_kernel <<<B_ * C_, 256, 0, stream>>>(in, pooled);
    se_gate_kernel <<<1, 128, 0, stream>>>(pooled, w1, b1, w2, b2, gate);
    se_scale_kernel<<<B_ * C_, 256, 0, stream>>>(in, gate, out);
}
